// Regression_82240033784362
// MI455X (gfx1250) — matrix operand order repaired and checked by reading, not compiled, not run on hardware
//
#include <hip/hip_runtime.h>
#include <hip/hip_bf16.h>

typedef _Float16 h8  __attribute__((ext_vector_type(8)));
typedef _Float16 h16 __attribute__((ext_vector_type(16)));
typedef float    f8  __attribute__((ext_vector_type(8)));

#define HID   256
#define DIN   39
#define DPAD  64
#define WAVES_PER_BLOCK 4
#define BLOCK_THREADS   (WAVES_PER_BLOCK * 32)
#define MAIN_BLOCKS     512

// ---- LDS layout (units: _Float16 elements) ----
#define W0T_OFF 0
#define W0T_ELE (HID * DPAD)            // 16384  (w0^T, [N=256][K=64], K-pad zeroed)
#define W1T_OFF (W0T_OFF + W0T_ELE)     // 16384
#define W1T_ELE (HID * HID)             // 65536  (w1^T, [N][K])
#define W2T_OFF (W1T_OFF + W1T_ELE)     // 81920
#define W2T_ELE (HID * HID)
#define WTS_ELE (W2T_OFF + W2T_ELE)     // 147456 halfs = 288 KB
#define ACT_OFF WTS_ELE
#define ACT_ELE (16 * HID)              // 4096 halfs = 8 KB per wave
#define LDS_HALFS (ACT_OFF + WAVES_PER_BLOCK * ACT_ELE)   // 163840
#define LDS_BYTES (LDS_HALFS * 2)                          // 327680 = 320 KB

// wave-local LDS ordering: CDNA5 split counter wait (cheaper than s_barrier,
// keeps the 4 waves of the block fully decoupled)
static __device__ __forceinline__ void lds_fence() {
  asm volatile("s_wait_dscnt 0" ::: "memory");
}

static __device__ __forceinline__ f8 wmma16(h16 a, h16 b, f8 c) {
  // v_wmma_f32_16x16x32_f16: (neg_a, A, neg_b, B, c_mod, C, reuse_a, reuse_b)
  return __builtin_amdgcn_wmma_f32_16x16x32_f16(false, a, false, b, (short)0, c,
                                                false, false);
}

// A fragment (16x32 f16, row-major source [16][stride] in LDS)
// lane<16: M=lane, elems 0..7 -> K=kb+0..7, elems 8..15 -> K=kb+16..23
// lane>=16: same M, K offset +8
static __device__ __forceinline__ h16 load_a(const _Float16* base, int stride,
                                             int kb, int lane) {
  const int m = lane & 15;
  const int hw = lane >> 4;
  const _Float16* p = base + m * stride + kb + hw * 8;
  h8 lo = *(const h8*)(p);
  h8 hi = *(const h8*)(p + 16);
  h16 r;
#pragma unroll
  for (int i = 0; i < 8; ++i) { r[i] = lo[i]; r[8 + i] = hi[i]; }
  return r;
}

// B fragment (32x16 f16) from W^T stored [N][K] (K contiguous) in LDS
// lane<16: N=nb+lane, elems 0..7 -> K=kb+0..7, elems 8..15 -> K=kb+16..23 ; lane>=16: K offset +8
static __device__ __forceinline__ h16 load_b(const _Float16* w, int stride,
                                             int nb, int kb, int lane) {
  const int nl = lane & 15;
  const int hw = lane >> 4;
  const _Float16* p = w + (nb + nl) * stride + kb + hw * 8;
  h8 lo = *(const h8*)(p);
  h8 hi = *(const h8*)(p + 16);
  h16 r;
#pragma unroll
  for (int i = 0; i < 8; ++i) { r[i] = lo[i]; r[8 + i] = hi[i]; }
  return r;
}

// D/C fragment: lane<16 holds N=lane, VGPR r = row M=r ; lane>=16: M=8+r
// bias folded here (C starts from inline 0 in the WMMAs)
static __device__ __forceinline__ void store_d_bias_relu(
    _Float16* act, int nb, f8 c, const float* __restrict__ bg, int lane) {
  const int col = nb + (lane & 15);
  const int mb  = (lane >> 4) * 8;
  const float bv = bg[col];
#pragma unroll
  for (int r = 0; r < 8; ++r) {
    float v = fmaxf(c[r] + bv, 0.f);
    act[(mb + r) * HID + col] = (_Float16)v;
  }
}

// ---------------- weight prep: f32 -> f16, transpose to [N][K] ----------------
__global__ void prep_weights(const float* __restrict__ w0,
                             const float* __restrict__ w1,
                             const float* __restrict__ w2,
                             _Float16* __restrict__ ws) {
  int idx = blockIdx.x * blockDim.x + threadIdx.x;
  if (idx >= WTS_ELE) return;
  float v;
  if (idx < W1T_OFF) {                       // w0t [256][64]
    int nn = idx >> 6, kk = idx & 63;
    v = (kk < DIN) ? w0[kk * HID + nn] : 0.f;
  } else if (idx < W2T_OFF) {                // w1t [256][256]
    int r = idx - W1T_OFF;
    int nn = r >> 8, kk = r & 255;
    v = w1[kk * HID + nn];
  } else {                                   // w2t [256][256]
    int r = idx - W2T_OFF;
    int nn = r >> 8, kk = r & 255;
    v = w2[kk * HID + nn];
  }
  ws[idx] = (_Float16)v;
}

// ---------------- fused transform + embed + MLP ----------------
__global__ __launch_bounds__(BLOCK_THREADS)
void nerfvq_main(const float* __restrict__ xw, const float* __restrict__ poses,
                 const _Float16* __restrict__ wts,
                 const float* __restrict__ b0g, const float* __restrict__ b1g,
                 const float* __restrict__ b2g, const float* __restrict__ w3g,
                 const float* __restrict__ b3g,
                 float* __restrict__ out_xla, float* __restrict__ out_ppt,
                 int bcount, int ncount, int ntiles) {
  extern __shared__ _Float16 smem[];

  // cooperative weight load: 288 KB global -> LDS as b128
  {
    const uint4* src = (const uint4*)wts;
    uint4* dst = (uint4*)smem;
    const int tot = (WTS_ELE * 2) / 16;
    for (int i = threadIdx.x; i < tot; i += blockDim.x) dst[i] = src[i];
  }
  __syncthreads();

  const int lane  = threadIdx.x & 31;
  const int wave  = threadIdx.x >> 5;
  const int m     = lane & 15;
  const int halfw = lane >> 4;
  _Float16* act = smem + ACT_OFF + wave * ACT_ELE;
  const _Float16* W0T = smem + W0T_OFF;
  const _Float16* W1T = smem + W1T_OFF;
  const _Float16* W2T = smem + W2T_OFF;

  const int gw = blockIdx.x * WAVES_PER_BLOCK + wave;
  const int gstride = gridDim.x * WAVES_PER_BLOCK;

  for (int tile = gw; tile < ntiles; tile += gstride) {
    lds_fence();  // previous tile's layer-3 reads complete before overwriting act

    const int grow0 = tile * 16;
    const int ni = grow0 / bcount;  // pose index, uniform across the tile
    float R[9], T[3];
#pragma unroll
    for (int r = 0; r < 3; ++r) {
#pragma unroll
      for (int c = 0; c < 3; ++c) R[r * 3 + c] = poses[ni * 16 + r * 4 + c];
      T[r] = poses[ni * 16 + r * 4 + 3];
    }
    const int grow = grow0 + m;
    const int bi = grow - ni * bcount;
    const float x0 = xw[bi * 3 + 0], x1 = xw[bi * 3 + 1], x2 = xw[bi * 3 + 2];
    float xl[3];
#pragma unroll
    for (int c = 0; c < 3; ++c)
      xl[c] = R[c * 3 + 0] * x0 + R[c * 3 + 1] * x1 + R[c * 3 + 2] * x2 + T[c];

    if (lane < 16) {
#pragma unroll
      for (int c = 0; c < 3; ++c) out_xla[grow * 3 + c] = xl[c];
      // NeRF positional encoding -> LDS row [m][0..63] (f16, zero padded)
      _Float16* row = act + m * DPAD;
      row[0] = (_Float16)xl[0];
      row[1] = (_Float16)xl[1];
      row[2] = (_Float16)xl[2];
      float f = 1.f;
#pragma unroll
      for (int i = 0; i < 6; ++i) {
#pragma unroll
        for (int c = 0; c < 3; ++c) {
          float a = xl[c] * f;
          row[3 + 6 * i + c]     = (_Float16)__sinf(a);
          row[3 + 6 * i + 3 + c] = (_Float16)__cosf(a);
        }
        f *= 2.f;
      }
#pragma unroll
      for (int kk = DIN; kk < DPAD; ++kk) row[kk] = (_Float16)0;
    }
    lds_fence();

    // ---- layer 0: [16x64] x w0t -> 16x256 ----
    {
      h16 a0 = load_a(act, DPAD, 0, lane);
      h16 a1 = load_a(act, DPAD, 32, lane);
#pragma unroll
      for (int hn = 0; hn < 2; ++hn) {
        f8 C[8] = {};  // inline-0 accumulator start
        // k-outer / n-inner: consecutive WMMAs hit different accumulators;
        // B fragment software-pipelined one step ahead
        h16 bf = load_b(W0T, DPAD, hn * 128, 0, lane);
#pragma unroll
        for (int step = 0; step < 16; ++step) {
          const int kk = step >> 3;
          const int nt = step & 7;
          h16 bn = bf;
          if (step < 15) {
            const int s = step + 1;
            bn = load_b(W0T, DPAD, hn * 128 + (s & 7) * 16, (s >> 3) * 32, lane);
          }
          C[nt] = wmma16(kk ? a1 : a0, bf, C[nt]);
          bf = bn;
        }
#pragma unroll
        for (int nt = 0; nt < 8; ++nt)
          store_d_bias_relu(act, hn * 128 + nt * 16, C[nt], b0g, lane);
      }
      lds_fence();
    }

    // ---- layers 1 & 2: 256x256 ----
#pragma unroll
    for (int L = 0; L < 2; ++L) {
      const _Float16* W = (L == 0) ? W1T : W2T;
      const float* bg = (L == 0) ? b1g : b2g;
      h16 A[8];
#pragma unroll
      for (int kk = 0; kk < 8; ++kk) A[kk] = load_a(act, HID, kk * 32, lane);
#pragma unroll
      for (int hn = 0; hn < 2; ++hn) {
        f8 C[8] = {};  // inline-0 accumulator start
        h16 bf = load_b(W, HID, hn * 128, 0, lane);
#pragma unroll
        for (int step = 0; step < 64; ++step) {
          const int kk = step >> 3;   // k-outer
          const int nt = step & 7;    // n-inner: independent WMMA back-to-back
          h16 bn = bf;
          if (step < 63) {
            const int s = step + 1;
            bn = load_b(W, HID, hn * 128 + (s & 7) * 16, (s >> 3) * 32, lane);
          }
          C[nt] = wmma16(A[kk], bf, C[nt]);
          bf = bn;
        }
#pragma unroll
        for (int nt = 0; nt < 8; ++nt)
          store_d_bias_relu(act, hn * 128 + nt * 16, C[nt], bg, lane);
      }
      lds_fence();
    }

    // ---- layer 3: 256 -> 1, VALU dot + xor-16 reduce ----
    {
      float sum = 0.f;
      const _Float16* row = act + m * HID + halfw * 128;
      const float* w3p = w3g + halfw * 128;
#pragma unroll
      for (int j = 0; j < 16; ++j) {
        h8 v = *(const h8*)(row + j * 8);
#pragma unroll
        for (int e = 0; e < 8; ++e) sum += (float)v[e] * w3p[j * 8 + e];
      }
      sum += __shfl_xor(sum, 16, 32);
      const float prob = sum + b3g[0];
      if (lane < 16) out_ppt[bi * ncount + ni] = prob;  // pred_prob.T slot
    }
  }
}

// ---------------- per-point argmin + gathers ----------------
__global__ void nerfvq_select(const float* __restrict__ xw,
                              const float* __restrict__ poses,
                              const float* __restrict__ vec,
                              const float* __restrict__ ppt,
                              float* __restrict__ out_xl,
                              float* __restrict__ out_me,
                              float* __restrict__ out_ipq,
                              float* __restrict__ out_pc,
                              float* __restrict__ out_vq,
                              int bcount, int ncount, int kdim) {
  int bi = blockIdx.x * blockDim.x + threadIdx.x;
  if (bi >= bcount) return;
  const float* p = ppt + (size_t)bi * ncount;
  int cls = 0;
  float best = p[0];
  for (int j = 1; j < ncount; ++j) {
    float v = p[j];
    if (v < best) { best = v; cls = j; }  // first-occurrence argmin
  }
  for (int j = 0; j < ncount; ++j)
    out_me[(size_t)bi * ncount + j] = (j == cls) ? 1.f : 0.f;
  float P[16];
#pragma unroll
  for (int i = 0; i < 16; ++i) {
    P[i] = poses[cls * 16 + i];
    out_ipq[(size_t)bi * 16 + i] = P[i];
  }
  const float x0 = xw[bi * 3], x1 = xw[bi * 3 + 1], x2 = xw[bi * 3 + 2];
#pragma unroll
  for (int c = 0; c < 3; ++c)
    out_xl[bi * 3 + c] =
        P[c * 4 + 0] * x0 + P[c * 4 + 1] * x1 + P[c * 4 + 2] * x2 + P[c * 4 + 3];
  out_pc[bi] = (float)cls;
  for (int i = 0; i < kdim; ++i)
    out_vq[(size_t)bi * kdim + i] = vec[cls * kdim + i];
}

extern "C" void kernel_launch(void* const* d_in, const int* in_sizes, int n_in,
                              void* d_out, int out_size, void* d_ws,
                              size_t ws_size, hipStream_t stream) {
  const float* xw    = (const float*)d_in[0];
  const float* poses = (const float*)d_in[1];
  const float* vec   = (const float*)d_in[2];
  const float* w0    = (const float*)d_in[3];
  const float* b0    = (const float*)d_in[4];
  const float* w1    = (const float*)d_in[5];
  const float* b1    = (const float*)d_in[6];
  const float* w2    = (const float*)d_in[7];
  const float* b2    = (const float*)d_in[8];
  const float* w3    = (const float*)d_in[9];
  const float* b3    = (const float*)d_in[10];

  const int bcount = in_sizes[0] / 3;        // 16384
  const int ncount = in_sizes[1] / 16;       // 32
  const int kdim   = in_sizes[2] / ncount;   // 8

  float* out = (float*)d_out;
  float* out_xl  = out;
  float* out_xla = out_xl + (size_t)bcount * 3;
  float* out_me  = out_xla + (size_t)ncount * bcount * 3;
  float* out_ipq = out_me + (size_t)bcount * ncount;
  float* out_pc  = out_ipq + (size_t)bcount * 16;
  float* out_ppt = out_pc + (size_t)bcount;
  float* out_vq  = out_ppt + (size_t)bcount * ncount;

  _Float16* wts = (_Float16*)d_ws;

  prep_weights<<<(WTS_ELE + 255) / 256, 256, 0, stream>>>(w0, w1, w2, wts);

  const int ntiles = (ncount * bcount) / 16;
  (void)hipFuncSetAttribute(reinterpret_cast<const void*>(nerfvq_main),
                            hipFuncAttributeMaxDynamicSharedMemorySize,
                            LDS_BYTES);
  nerfvq_main<<<MAIN_BLOCKS, BLOCK_THREADS, LDS_BYTES, stream>>>(
      xw, poses, wts, b0, b1, b2, w3, b3, out_xla, out_ppt, bcount, ncount,
      ntiles);
  nerfvq_select<<<(bcount + 255) / 256, 256, 0, stream>>>(
      xw, poses, vec, out_ppt, out_xl, out_me, out_ipq, out_pc, out_vq, bcount,
      ncount, kdim);
}
